// AttentionModule_61272003444953
// MI455X (gfx1250) — matrix operand order repaired and checked by reading, not compiled, not run on hardware
//
#include <hip/hip_runtime.h>
#include <hip/hip_bf16.h>
#include <math.h>

// Problem constants (S,B,H) = (2048, 32, 1024)
#define S_LEN 2048
#define B_SZ  32
#define H_SZ  1024

typedef __attribute__((ext_vector_type(16))) __bf16 v16bf;
typedef __attribute__((ext_vector_type(8)))  __bf16 v8bf;
typedef __attribute__((ext_vector_type(4)))  __bf16 v4bf;
typedef __attribute__((ext_vector_type(8)))  float  v8f;

// ---------------------------------------------------------------------------
// Branchless tanh. Prefer gfx1250 hardware v_tanh_f32 if the builtin exists;
// otherwise use exp-based identity (v_exp_f32 + v_rcp_f32, no divergence):
//   tanh(x) = sign(x) * (1 - e^(-2|x|)) / (1 + e^(-2|x|))
// ---------------------------------------------------------------------------
#if defined(__has_builtin)
#if __has_builtin(__builtin_amdgcn_tanhf)
#define HW_TANH(x) __builtin_amdgcn_tanhf(x)
#elif __has_builtin(__builtin_amdgcn_tanh_f32)
#define HW_TANH(x) __builtin_amdgcn_tanh_f32(x)
#endif
#endif

__device__ __forceinline__ float fast_tanh(float x) {
#ifdef HW_TANH
    return HW_TANH(x);
#else
    float ax = __builtin_fabsf(x);
    float e  = __expf(-2.0f * ax);                       // native v_exp_f32 path
    float r  = (1.0f - e) * __builtin_amdgcn_rcpf(1.0f + e);
    return __builtin_copysignf(r, x);
#endif
}

// ---------------------------------------------------------------------------
// Kernel 0: convert encoder-half of attn_w to bf16, stored column-contiguous:
//   Wb[n*1024 + k] = bf16( attn_w[n*2H + H + k] )   (B-matrix: K contiguous per column)
// ---------------------------------------------------------------------------
__global__ void convert_w_kernel(const float* __restrict__ attn_w,
                                 __bf16* __restrict__ Wb) {
    int idx = (blockIdx.x * 256 + threadIdx.x) * 4;   // 0 .. 1M-4
    int n = idx >> 10;
    int k = idx & 1023;
    float4 f = *(const float4*)(attn_w + (size_t)n * 2048 + 1024 + k);
    v4bf o;
    o[0] = (__bf16)f.x; o[1] = (__bf16)f.y; o[2] = (__bf16)f.z; o[3] = (__bf16)f.w;
    *(v4bf*)(Wb + idx) = o;
}

// ---------------------------------------------------------------------------
// Kernel 1: pre[b*H + h] = attn_b[h] + dot(hidden[b,:], attn_w[h, 0:H])
// 32x1024 outputs, K=1024 — tiny (67 MFLOP), plain VALU.
// ---------------------------------------------------------------------------
__global__ void pre_kernel(const float* __restrict__ hidden,
                           const float* __restrict__ attn_w,
                           const float* __restrict__ attn_b,
                           float* __restrict__ pre) {
    int gid = blockIdx.x * 256 + threadIdx.x;   // 0..32767
    int b = gid >> 10, h = gid & 1023;
    const float4* wr = (const float4*)(attn_w + (size_t)h * 2048);
    const float4* hr = (const float4*)(hidden + (size_t)b * 1024);
    float acc = attn_b[h];
#pragma unroll 4
    for (int k = 0; k < 256; ++k) {
        float4 wv = wr[k];
        float4 hv = hr[k];
        acc += wv.x * hv.x + wv.y * hv.y + wv.z * hv.z + wv.w * hv.w;
    }
    pre[gid] = acc;
}

// ---------------------------------------------------------------------------
// Kernel 2: fused bf16-WMMA GEMM + tanh + v-dot.
// Rows r = s*B + b of enc (contiguous, length H). M=65536, N=1024, K=1024.
// 1024 blocks x 128 threads (4 waves). Per block: 64-row A tile staged f32->bf16
// in 128KB LDS; each wave computes 16 rows over all N, 16 cols per chunk,
// 32x v_wmma_f32_16x16x32_bf16 per chunk, epilogue reduces with v into scores.
// ---------------------------------------------------------------------------
__global__ void energy_kernel(const float* __restrict__ enc,
                              const __bf16* __restrict__ Wb,
                              const float* __restrict__ pre,
                              const float* __restrict__ vvec,
                              float* __restrict__ scores) {
    __shared__ __bf16 ldsA[64 * 1024];          // 128 KB

    const int tid = threadIdx.x;                 // 0..127
    const size_t wgBase = (size_t)blockIdx.x * 64;

    // ---- stage contiguous 64x1024 f32 A-tile into LDS as bf16 ----
    const float4* src = (const float4*)(enc + wgBase * 1024);
#pragma unroll 4
    for (int it = 0; it < 128; ++it) {
        int c = it * 128 + tid;                  // 0..16383 float4 chunks
        float4 f = src[c];
        v4bf o;
        o[0] = (__bf16)f.x; o[1] = (__bf16)f.y; o[2] = (__bf16)f.z; o[3] = (__bf16)f.w;
        *(v4bf*)&ldsA[c * 4] = o;
    }
    __syncthreads();

    const int wave = tid >> 5;
    const int lane = tid & 31;
    const int half = lane >> 4;                  // 0: lanes 0-15, 1: lanes 16-31
    const int lcol = lane & 15;
    const int rowBase = wave * 16;               // this wave's 16 rows in tile

    // A operand row for this lane (ISA 16-bit A layout: M = lane & 15)
    const __bf16* arow = &ldsA[(size_t)(rowBase + lcol) * 1024];

    float acc[8] = {0.f, 0.f, 0.f, 0.f, 0.f, 0.f, 0.f, 0.f};

    for (int n0 = 0; n0 < 1024; n0 += 16) {
        const int n = n0 + lcol;                 // this lane's output column
        const __bf16* bcol = Wb + (size_t)n * 1024;  // K-contiguous column
        v8f c = {};
#pragma unroll 4
        for (int k0 = 0; k0 < 1024; k0 += 32) {
            // A 16x32 bf16: lane<16 holds K {k0+0..7, k0+16..23}; lane>=16 K {+8..15, +24..31}
            v8bf alo = *(const v8bf*)(arow + k0 + half * 8);
            v8bf ahi = *(const v8bf*)(arow + k0 + 16 + half * 8);
            v16bf a = __builtin_shufflevector(alo, ahi,
                    0, 1, 2, 3, 4, 5, 6, 7, 8, 9, 10, 11, 12, 13, 14, 15);
            // B 32x16 bf16: lane<16 holds col n, K {k0+0..7, k0+16..23}; lane>=16 K {+8..15, +24..31}
            v8bf blo = *(const v8bf*)(bcol + k0 + half * 8);
            v8bf bhi = *(const v8bf*)(bcol + k0 + 16 + half * 8);
            v16bf bm = __builtin_shufflevector(blo, bhi,
                    0, 1, 2, 3, 4, 5, 6, 7, 8, 9, 10, 11, 12, 13, 14, 15);
            c = __builtin_amdgcn_wmma_f32_16x16x32_bf16(
                    false, a, false, bm, (short)0, c, false, false);
        }
        // Epilogue: C VGPR j holds row M=j (lanes 0-15) / M=8+j (lanes 16-31),
        // column N = lane&15 within this chunk. Branchless tanh keeps VALU
        // pressure low so WMMA (TRANS-class) co-execution isn't starved.
        const float vn = vvec[n];
        const int mrow = rowBase + half * 8;
#pragma unroll
        for (int j = 0; j < 8; ++j) {
            size_t gm = wgBase + mrow + j;                 // global row r = s*B + b
            float e = fast_tanh(c[j] + pre[(gm & 31) * 1024 + n]);
            acc[j] = fmaf(e, vn, acc[j]);
        }
    }

    // Reduce each row's partial score across the 16 lanes of its half.
#pragma unroll
    for (int j = 0; j < 8; ++j) {
        float aj = acc[j];
        aj += __shfl_xor(aj, 1);
        aj += __shfl_xor(aj, 2);
        aj += __shfl_xor(aj, 4);
        aj += __shfl_xor(aj, 8);
        acc[j] = aj;
    }
    if (lcol == 0) {
#pragma unroll
        for (int j = 0; j < 8; ++j) {
            size_t gm = wgBase + rowBase + half * 8 + j;   // r = s*B + b
            scores[(gm & 31) * S_LEN + (gm >> 5)] = acc[j];
        }
    }
}

// ---------------------------------------------------------------------------
// Kernel 3: masked softmax over S per batch row; writes attention weights.
// ---------------------------------------------------------------------------
__global__ void softmax_kernel(const float* __restrict__ scores,
                               const int* __restrict__ lens,
                               float* __restrict__ weights) {
    __shared__ float red[256];
    const int b = blockIdx.x;
    const int tid = threadIdx.x;
    const int len = lens[b];

    float vals[8];
    float mx = -3.4e38f;
#pragma unroll
    for (int i = 0; i < 8; ++i) {
        int s = tid + i * 256;
        float x = scores[b * S_LEN + s];
        if (s >= len) x = -1e10f;
        vals[i] = x;
        mx = fmaxf(mx, x);
    }
    red[tid] = mx; __syncthreads();
    for (int off = 128; off > 0; off >>= 1) {
        if (tid < off) red[tid] = fmaxf(red[tid], red[tid + off]);
        __syncthreads();
    }
    mx = red[0]; __syncthreads();

    float sum = 0.f;
#pragma unroll
    for (int i = 0; i < 8; ++i) {
        vals[i] = expf(vals[i] - mx);
        sum += vals[i];
    }
    red[tid] = sum; __syncthreads();
    for (int off = 128; off > 0; off >>= 1) {
        if (tid < off) red[tid] += red[tid + off];
        __syncthreads();
    }
    const float inv = 1.f / red[0];
#pragma unroll
    for (int i = 0; i < 8; ++i)
        weights[b * S_LEN + tid + i * 256] = vals[i] * inv;
}

// ---------------------------------------------------------------------------
// Kernel 4: context[b,h] = sum_s w[b,s] * enc[s,b,h]  (memory bound, 256 MB read)
// grid (B, H/256, 16 s-chunks of 128); float atomics accumulate partials.
// ---------------------------------------------------------------------------
__global__ void context_kernel(const float* __restrict__ w,
                               const float* __restrict__ enc,
                               float* __restrict__ ctx) {
    const int b = blockIdx.x;
    const int h = blockIdx.y * 256 + threadIdx.x;
    const int s0 = blockIdx.z * 128;
    const float* encp = enc + ((size_t)s0 * B_SZ + b) * 1024 + h;
    float acc = 0.f;
#pragma unroll 4
    for (int i = 0; i < 128; ++i)
        acc += w[b * S_LEN + s0 + i] * encp[(size_t)i * B_SZ * 1024];
    atomicAdd(&ctx[b * 1024 + h], acc);
}

// ---------------------------------------------------------------------------
extern "C" void kernel_launch(void* const* d_in, const int* in_sizes, int n_in,
                              void* d_out, int out_size, void* d_ws, size_t ws_size,
                              hipStream_t stream) {
    const float* hidden = (const float*)d_in[0];   // (B,H)
    const float* enc    = (const float*)d_in[1];   // (S,B,H)
    const int*   lens   = (const int*)  d_in[2];   // (B,)
    const float* attn_w = (const float*)d_in[3];   // (H,2H)
    const float* attn_b = (const float*)d_in[4];   // (H,)
    const float* vvec   = (const float*)d_in[5];   // (H,)
    float* out = (float*)d_out;                    // context (B*H) ++ weights (B*S)

    char* ws = (char*)d_ws;
    __bf16* Wb    = (__bf16*)ws;                                   // 2 MB: W2^T bf16
    float*  pre   = (float*)(ws + 2 * 1024 * 1024);                // 128 KB
    float*  score = (float*)(ws + 2 * 1024 * 1024 + 128 * 1024);   // 256 KB

    float* weights = out + B_SZ * H_SZ;

    convert_w_kernel<<<1024, 256, 0, stream>>>(attn_w, Wb);
    pre_kernel<<<128, 256, 0, stream>>>(hidden, attn_w, attn_b, pre);
    energy_kernel<<<(S_LEN * B_SZ) / 64, 128, 0, stream>>>(enc, Wb, pre, vvec, score);
    softmax_kernel<<<B_SZ, 256, 0, stream>>>(score, lens, weights);
    hipMemsetAsync(out, 0, (size_t)B_SZ * H_SZ * sizeof(float), stream);
    context_kernel<<<dim3(B_SZ, H_SZ / 256, 16), 256, 0, stream>>>(weights, enc, out);
}
